// BasicBlock_1w8a_q_13804024889764
// MI455X (gfx1250) — 8-bit matrix operand order repaired and checked by reading, compiled, not run on hardware
//
#include <hip/hip_runtime.h>
#include <hip/hip_bf16.h>
#include <stdint.h>

typedef __attribute__((ext_vector_type(8))) int v8i;

#define B_    128
#define C_    64
#define H_    56
#define W_    56
#define HW_   (H_ * W_)
#define NPIX  (B_ * HW_)
#define NEL   ((size_t)B_ * C_ * HW_)

#define WS_MAXBITS   0
#define WS_SW        256
#define WS_QW        768
#define WS_WPACK     2048
#define WPACK_DW     (4 * 9 * 32 * 8)
#define WS_A1        131072
#define WS_A2        (WS_A1 + 25690112)
#define WS_MID       (WS_A2 + 25690112)

__global__ void k_init(unsigned* maxbits) {
    if (threadIdx.x < 2) maxbits[threadIdx.x] = 0u;
}

__global__ __launch_bounds__(256) void k_absmax(const float4* __restrict__ x,
                                                unsigned* maxbits, long n4) {
    __shared__ float red[256];
    float m = 0.f;
    for (long i = (long)blockIdx.x * blockDim.x + threadIdx.x; i < n4;
         i += (long)gridDim.x * blockDim.x) {
        float4 v = x[i];
        m = fmaxf(m, fmaxf(fmaxf(fabsf(v.x), fabsf(v.y)),
                           fmaxf(fabsf(v.z), fabsf(v.w))));
    }
    red[threadIdx.x] = m;
    __syncthreads();
    for (int s = 128; s > 0; s >>= 1) {
        if (threadIdx.x < s) red[threadIdx.x] = fmaxf(red[threadIdx.x], red[threadIdx.x + s]);
        __syncthreads();
    }
    if (threadIdx.x == 0) atomicMax(maxbits, __float_as_uint(red[0]));
}

__global__ __launch_bounds__(64) void k_wprep(const float* __restrict__ w1,
                                              const float* __restrict__ w2,
                                              int* __restrict__ wpack_base,
                                              float* __restrict__ sw_base) {
    const int co = blockIdx.x;
    const int wsel = blockIdx.y;
    const float* w = (wsel ? w2 : w1) + (size_t)co * 576;
    int* wpack = wpack_base + wsel * WPACK_DW;
    float* sw = sw_base + wsel * 64;

    __shared__ float red[64], red2[64];
    __shared__ float s_mean, s_rstd;
    const int t = threadIdx.x;

    float s = 0.f, sq = 0.f;
    for (int i = t; i < 576; i += 64) { float v = w[i]; s += v; sq += v * v; }
    red[t] = s; red2[t] = sq;
    __syncthreads();
    for (int k = 32; k > 0; k >>= 1) {
        if (t < k) { red[t] += red[t + k]; red2[t] += red2[t + k]; }
        __syncthreads();
    }
    if (t == 0) {
        float mean = red[0] * (1.f / 576.f);
        float var = (red2[0] - 576.f * mean * mean) * (1.f / 575.f);
        s_mean = mean;
        s_rstd = 1.f / sqrtf(var);
    }
    __syncthreads();
    const float mean = s_mean;

    float aab = 0.f;
    for (int i = t; i < 576; i += 64) aab += fabsf(w[i] - mean);
    red[t] = aab;
    __syncthreads();
    for (int k = 32; k > 0; k >>= 1) {
        if (t < k) red[t] += red[t + k];
        __syncthreads();
    }
    if (t == 0) {
        float mabs = red[0] * (1.f / 576.f) * s_rstd;
        sw[co] = exp2f(rintf(log2f(mabs)));
    }

    const int n = co & 15, nt = co >> 4;
    for (int d = t; d < 144; d += 64) {
        int tap = d >> 4, slot = d & 15;
        int lane = n + ((slot >= 8) ? 16 : 0);
        int vgpr = slot & 7;
        int k0 = ((vgpr & 1) * 4) + (((vgpr >> 1) & 1) * 16) + (((vgpr >> 2) & 1) * 32) + ((lane >= 16) ? 8 : 0);
        int kh = tap / 3, kw = tap % 3;
        unsigned dw = 0;
        #pragma unroll
        for (int j = 0; j < 4; ++j) {
            int ci = k0 + j;
            float v = w[ci * 9 + kh * 3 + kw] - mean;
            unsigned byte = (v >= 0.f) ? 0x01u : 0xFFu;
            dw |= byte << (8 * j);
        }
        wpack[((nt * 9 + tap) * 32 + lane) * 8 + vgpr] = (int)dw;
    }
}

__global__ __launch_bounds__(64) void k_bnprep(const float* g1, const float* bb1,
                                               const float* mm1, const float* vv1,
                                               const float* g2, const float* bb2,
                                               const float* mm2, const float* vv2,
                                               float* qwb) {
    const int sel = blockIdx.x;
    const float* g  = sel ? g2  : g1;
    const float* be = sel ? bb2 : bb1;
    const float* mu = sel ? mm2 : mm1;
    const float* va = sel ? vv2 : vv1;
    float* qw = qwb + sel * 128;
    float* qb = qw + 64;
    const int t = threadIdx.x;

    float wv = g[t] / sqrtf(va[t] + 1e-5f);
    float bv = be[t] - wv * mu[t];

    __shared__ float red[64];
    red[t] = fabsf(wv);
    __syncthreads();
    for (int k = 32; k > 0; k >>= 1) {
        if (t < k) red[t] = fmaxf(red[t], red[t + k]);
        __syncthreads();
    }
    float Tw = fminf(fmaxf(red[0], 1e-10f), 255.f);
    __syncthreads();
    red[t] = fabsf(bv);
    __syncthreads();
    for (int k = 32; k > 0; k >>= 1) {
        if (t < k) red[t] = fmaxf(red[t], red[t + k]);
        __syncthreads();
    }
    float Tb = fminf(fmaxf(red[0], 1e-10f), 255.f);

    qw[t] = rintf(wv / Tw * 7.f)    * (1.f / 7.f)    * Tw;
    qb[t] = rintf(bv / Tb * 4095.f) * (1.f / 4095.f) * Tb;
}

__global__ __launch_bounds__(256) void k_quant(const float* __restrict__ src,
                                               unsigned* __restrict__ dst,
                                               const unsigned* __restrict__ maxbits) {
    float T = __uint_as_float(*maxbits);
    T = fminf(fmaxf(T, 1e-10f), 255.f);
    const float s7 = 7.f / T;
    const int p = blockIdx.x * blockDim.x + threadIdx.x;
    if (p >= NPIX) return;
    const int b = p / HW_, hw = p % HW_;
    const float* base = src + (size_t)b * C_ * HW_ + hw;
    #pragma unroll 4
    for (int cg = 0; cg < 16; ++cg) {
        unsigned dw = 0;
        #pragma unroll
        for (int j = 0; j < 4; ++j) {
            int ci = cg * 4 + j;
            float v = base[(size_t)ci * HW_];
            float q = rintf(fminf(fmaxf(v, -T), T) * s7);
            dw |= (((unsigned)(int)q) & 0xFFu) << (8 * j);
        }
        dst[(size_t)p * 16 + cg] = dw;
    }
}

__global__ __launch_bounds__(128) void k_conv(const unsigned* __restrict__ act,
                                              const int* __restrict__ wpack,
                                              const float* __restrict__ sw,
                                              const float* __restrict__ qw,
                                              const float* __restrict__ qb,
                                              float* __restrict__ out,
                                              unsigned* t2bits, int do_t2) {
    __shared__ signed char smem[4 * 66 * 64];
    __shared__ int wlds[WPACK_DW];
    __shared__ float redm[128];
    const int h0 = blockIdx.x * 2, b = blockIdx.y;
    const int tid = threadIdx.x;

    __builtin_prefetch(act, 0, 3);

    for (int d = tid; d < WPACK_DW; d += 128) wlds[d] = wpack[d];

    unsigned* smem_u = (unsigned*)smem;
    for (int d = tid; d < 4 * 66 * 16; d += 128) {
        int r = d / (66 * 16);
        int rem = d - r * (66 * 16);
        int wco = rem >> 4;
        int cg = rem & 15;
        int hin = h0 + r - 1;
        int win = wco - 1;
        unsigned val = 0;
        if (hin >= 0 && hin < H_ && win >= 0 && win < W_) {
            size_t pix = (size_t)(b * H_ + hin) * W_ + win;
            val = act[pix * 16 + cg];
        }
        smem_u[d] = val;
    }
    __syncthreads();

    const int wave = tid >> 5, lane = tid & 31;
    const int mrow = lane & 15;
    const int kbase = (lane >> 4) * 8;
    const int mpix = wave * 16 + mrow;

    v8i acc[2][4];
    #pragma unroll
    for (int r = 0; r < 2; ++r)
        #pragma unroll
        for (int nt = 0; nt < 4; ++nt) acc[r][nt] = (v8i){0, 0, 0, 0, 0, 0, 0, 0};

    #pragma unroll
    for (int tap = 0; tap < 9; ++tap) {
        const int kh = tap / 3, kw = tap % 3;
        const int ab0 = ((kh)     * 66 + (mpix + kw)) * 64;
        const int ab1 = ((kh + 1) * 66 + (mpix + kw)) * 64;
        v8i a0, a1;
        #pragma unroll
        for (int v = 0; v < 8; ++v) {
            int koff = ((v & 1) * 4) + (((v >> 1) & 1) * 16) + (((v >> 2) & 1) * 32) + kbase;
            a0[v] = *(const int*)(&smem[ab0 + koff]);
            a1[v] = *(const int*)(&smem[ab1 + koff]);
        }
        #pragma unroll
        for (int nt = 0; nt < 4; ++nt) {
            v8i bf = *(const v8i*)(&wlds[((nt * 9 + tap) * 32 + lane) * 8]);
            acc[0][nt] = __builtin_amdgcn_wmma_i32_16x16x64_iu8(
                true, a0, true, bf, acc[0][nt], false, false);
            acc[1][nt] = __builtin_amdgcn_wmma_i32_16x16x64_iu8(
                true, a1, true, bf, acc[1][nt], false, false);
        }
    }

    float lmax = 0.f;
    const int ncol = lane & 15;
    #pragma unroll
    for (int r = 0; r < 2; ++r) {
        const int h = h0 + r;
        #pragma unroll
        for (int nt = 0; nt < 4; ++nt) {
            const int n = nt * 16 + ncol;
            const float swn = sw[n] * (1.f / 7.f);
            const float qwn = qw[n], qbn = qb[n];
            #pragma unroll
            for (int v = 0; v < 8; ++v) {
                int m = (lane < 16) ? v : v + 8;
                int wpix = wave * 16 + m;
                if (wpix < W_) {
                    float conv = (float)acc[r][nt][v] * swn;
                    float aq = rintf(conv * (1023.f / 576.f)) * (576.f / 1023.f);
                    float scv = (float)smem[((1 + r) * 66 + (wpix + 1)) * 64 + n] * (1.f / 7.f);
                    float o = aq * qwn + qbn + scv;
                    o = fminf(1.f, fmaxf(-1.f, o));
                    out[(((size_t)b * C_ + n) * H_ + h) * W_ + wpix] = o;
                    lmax = fmaxf(lmax, fabsf(o));
                }
            }
        }
    }

    if (do_t2) {
        redm[tid] = lmax;
        __syncthreads();
        for (int s = 64; s > 0; s >>= 1) {
            if (tid < s) redm[tid] = fmaxf(redm[tid], redm[tid + s]);
            __syncthreads();
        }
        if (tid == 0) atomicMax(t2bits, __float_as_uint(redm[0]));
    }
}

extern "C" void kernel_launch(void* const* d_in, const int* in_sizes, int n_in,
                              void* d_out, int out_size, void* d_ws, size_t ws_size,
                              hipStream_t stream) {
    const float* x  = (const float*)d_in[0];
    const float* w1 = (const float*)d_in[1];
    const float* w2 = (const float*)d_in[2];
    const float* g1 = (const float*)d_in[3];
    const float* b1 = (const float*)d_in[4];
    const float* m1 = (const float*)d_in[5];
    const float* v1 = (const float*)d_in[6];
    const float* g2 = (const float*)d_in[7];
    const float* b2 = (const float*)d_in[8];
    const float* m2 = (const float*)d_in[9];
    const float* v2 = (const float*)d_in[10];
    float* out = (float*)d_out;

    char* W = (char*)d_ws;
    unsigned* maxbits = (unsigned*)(W + WS_MAXBITS);
    float* sw_base = (float*)(W + WS_SW);
    float* qw_base = (float*)(W + WS_QW);
    int* wpack1 = (int*)(W + WS_WPACK);
    int* wpack2 = wpack1 + WPACK_DW;
    unsigned* a1 = (unsigned*)(W + WS_A1);
    unsigned* a2 = (unsigned*)(W + WS_A2);
    float* mid = (float*)(W + WS_MID);

    k_init<<<1, 32, 0, stream>>>(maxbits);
    k_wprep<<<dim3(64, 2), 64, 0, stream>>>(w1, w2, wpack1, sw_base);
    k_bnprep<<<2, 64, 0, stream>>>(g1, b1, m1, v1, g2, b2, m2, v2, qw_base);

    k_absmax<<<2048, 256, 0, stream>>>((const float4*)x, maxbits + 0, (long)(NEL / 4));

    k_quant<<<(NPIX + 255) / 256, 256, 0, stream>>>(x, a1, maxbits + 0);

    k_conv<<<dim3(H_ / 2, B_), 128, 0, stream>>>(a1, wpack1, sw_base, qw_base,
                                                 qw_base + 64, mid, maxbits + 1, 1);

    k_quant<<<(NPIX + 255) / 256, 256, 0, stream>>>(mid, a2, maxbits + 1);

    k_conv<<<dim3(H_ / 2, B_), 128, 0, stream>>>(a2, wpack2, sw_base + 64, qw_base + 128,
                                                 qw_base + 192, out, nullptr, 0);
}
